// CrossNetMix_47820165874255
// MI455X (gfx1250) — matrix operand order repaired and checked by reading, not compiled, not run on hardware
//
#include <hip/hip_runtime.h>
#include <hip/hip_bf16.h>

// Problem constants (match reference)
#define BB 16384
#define DD 1024
#define RR 64
#define EE 4
#define LL 3

typedef __attribute__((ext_vector_type(16))) __bf16 v16bf;
typedef __attribute__((ext_vector_type(8)))  float  v8f;

// ---------------------------------------------------------------------------
// Fragment loaders: A and B share contiguous k order (k = 16h + s); C/D per CDNA5 layout
// A (16x32 bf16): lane l -> row m = l&15 ; K-half kh = (l>>4)*16
//   elements 0..15 = K kh..kh+15 (contiguous)
// B (32x16 bf16, stored transposed Bt[n][k]): lane l -> col n = l&15,
//   K range = kb + (l>=16 ? 16 : 0), 16 contiguous bf16 (32B aligned)
// C/D (16x16 f32): VGPR v, lanes 0-15: M=v, N=lane; lanes 16-31: M=v+8
// ---------------------------------------------------------------------------
__device__ __forceinline__ v16bf load_a_global_f32(const float* __restrict__ x,
                                                   int rowbase, int kb, int lane) {
  int m  = lane & 15;
  int kh = (lane >> 4) << 4;
  const float* p = x + (rowbase + m) * DD + kb + kh;
  v16bf a;
#pragma unroll
  for (int i = 0; i < 8; ++i) a[i] = (__bf16)p[i];
#pragma unroll
  for (int i = 0; i < 8; ++i) a[8 + i] = (__bf16)p[8 + i];
  return a;
}

__device__ __forceinline__ v16bf load_a_lds(const __bf16* s, int ld, int kb, int lane) {
  int m  = lane & 15;
  int kh = (lane >> 4) << 4;
  const __bf16* p = s + m * ld + kb + kh;
  v16bf a;
#pragma unroll
  for (int i = 0; i < 8; ++i) a[i] = p[i];
#pragma unroll
  for (int i = 0; i < 8; ++i) a[8 + i] = p[8 + i];
  return a;
}

__device__ __forceinline__ v16bf load_b_tr(const __bf16* __restrict__ bt, int ld,
                                           int nbase, int kb, int lane) {
  int n  = lane & 15;
  int k0 = kb + ((lane >> 4) << 4);
  return *(const v16bf*)(bt + (nbase + n) * ld + k0);
}

__device__ __forceinline__ v8f wmma_bf16(v16bf a, v16bf b, v8f c) {
  return __builtin_amdgcn_wmma_f32_16x16x32_bf16(false, a, false, b, (short)0, c,
                                                 false, false);
}

__device__ __forceinline__ float fast_tanhf(float x) {
  x = fminf(fmaxf(x, -15.0f), 15.0f);
  float e = __expf(2.0f * x);
  return (e - 1.0f) / (e + 1.0f);
}

// ---------------------------------------------------------------------------
// Weight prep: bf16 conversion + transposes so WMMA B-fragments are contiguous
//   Vt[l][e*64+r][d] = V[l,e,d,r]   (ld = 1024)
//   Ut[l][d][e*64+r] = U[l,e,d,r]   (ld = 256)
//   Ct[l][e][r][s]   = C[l,e,r,s]   (ld = 64)  (Bt[n=r][k=s] is exactly C row-major)
//   Gt[j][d] = j<4 ? gate_w[j,d] : 0  (16-col padded gate tile, ld = 1024)
// ---------------------------------------------------------------------------
__global__ void crossnet_prep_kernel(const float* __restrict__ U,
                                     const float* __restrict__ V,
                                     const float* __restrict__ C,
                                     const float* __restrict__ gw,
                                     __bf16* __restrict__ Vt,
                                     __bf16* __restrict__ Ut,
                                     __bf16* __restrict__ Ct,
                                     __bf16* __restrict__ Gt) {
  int idx = blockIdx.x * blockDim.x + threadIdx.x;
  const int nUV = LL * EE * DD * RR;  // 786432
  if (idx < nUV) {
    int r = idx % RR;
    int d = (idx / RR) % DD;
    int e = (idx / (RR * DD)) % EE;
    int l = idx / (RR * DD * EE);
    Vt[(l * (EE * RR) + e * RR + r) * DD + d] = (__bf16)V[idx];
    Ut[(l * DD + d) * (EE * RR) + e * RR + r] = (__bf16)U[idx];
  }
  if (idx < LL * EE * RR * RR) Ct[idx] = (__bf16)C[idx];
  if (idx < 16 * DD) {
    int j = idx >> 10, d = idx & (DD - 1);
    Gt[idx] = (j < EE) ? (__bf16)gw[j * DD + d] : (__bf16)0.0f;
  }
}

// ---------------------------------------------------------------------------
// GEMM1 inner loop, software-pipelined and unrolled x2 so the scheduler gets
// two live A/B register sets: the cvt building A for step k+1 overlaps the
// wmma reading A of step k without WMMA->VALU WAR hazard nops. GATE path
// (wave 0) handled by a hoisted uniform branch: EXEC is all-ones at every
// wmma site, no per-iteration saveexec.
// ---------------------------------------------------------------------------
template <bool GATE>
__device__ __forceinline__ void gemm1_compute(const float* __restrict__ xin,
                                              const __bf16* __restrict__ Vt,
                                              const __bf16* __restrict__ Gt,
                                              int rowbase, int t0, int t1, int lane,
                                              v8f& acc0, v8f& acc1, v8f& accG) {
  v16bf a  = load_a_global_f32(xin, rowbase, 0, lane);
  v16bf b0 = load_b_tr(Vt, DD, t0 * 16, 0, lane);
  v16bf b1 = load_b_tr(Vt, DD, t1 * 16, 0, lane);
  v16bf bg{};
  if (GATE) bg = load_b_tr(Gt, DD, 0, 0, lane);
#pragma unroll 2
  for (int kk = 0; kk < 31; ++kk) {
    const int kb = (kk + 1) * 32;
    v16bf an  = load_a_global_f32(xin, rowbase, kb, lane);
    v16bf b0n = load_b_tr(Vt, DD, t0 * 16, kb, lane);
    v16bf b1n = load_b_tr(Vt, DD, t1 * 16, kb, lane);
    v16bf bgn{};
    if (GATE) bgn = load_b_tr(Gt, DD, 0, kb, lane);
    acc0 = wmma_bf16(a, b0, acc0);
    acc1 = wmma_bf16(a, b1, acc1);
    if (GATE) accG = wmma_bf16(a, bg, accG);
    a = an; b0 = b0n; b1 = b1n;
    if (GATE) bg = bgn;
  }
  acc0 = wmma_bf16(a, b0, acc0);
  acc1 = wmma_bf16(a, b1, acc1);
  if (GATE) accG = wmma_bf16(a, bg, accG);
}

// ---------------------------------------------------------------------------
// Fused layer kernel: 16 rows per block, 256 threads = 8 waves.
// Phase 1: Y1[16,256] = x_l @ Vb (16 N-tiles) + gate logits (1 padded tile)
// Phase 2: softmax over 4 experts (per-row)
// Phase 3: v2 = tanh(v1 @ C_e^T) via WMMA, scaled by gate -> w (bf16 LDS)
// Phase 4: out = x0 * (w @ U^T + bias) + x_l  (64 N-tiles, K=256)
// ---------------------------------------------------------------------------
#define LDPAD 264

__global__ __launch_bounds__(256) void crossnet_layer_kernel(
    const float* __restrict__ xin, const float* __restrict__ x0,
    const __bf16* __restrict__ Vt, const __bf16* __restrict__ Gt,
    const __bf16* __restrict__ Ct, const __bf16* __restrict__ Ut,
    const float* __restrict__ bias, float* __restrict__ xout) {
  __shared__ __bf16 sV1[16 * LDPAD];
  __shared__ __bf16 sW[16 * LDPAD];
  __shared__ float sLog[16][EE];
  __shared__ float sG[16][EE];

  const int lane = threadIdx.x & 31;
  const int wave = threadIdx.x >> 5;
  const int rowbase = blockIdx.x * 16;
  const int mrow = ((lane >> 4) << 3);  // C/D: +0 or +8 row offset per half
  const int ncol = lane & 15;

  // ---- Phase 1: GEMM1 (x_l @ V concat) + gate logits -----------------------
  {
    const int t0 = wave;
    const int t1 = wave + 8;
    v8f acc0{}, acc1{}, accG{};
    if (wave == 0) {  // wave-uniform branch, hoisted out of the k-loop
      gemm1_compute<true>(xin, Vt, Gt, rowbase, t0, t1, lane, acc0, acc1, accG);
    } else {
      gemm1_compute<false>(xin, Vt, Gt, rowbase, t0, t1, lane, acc0, acc1, accG);
    }
#pragma unroll
    for (int v = 0; v < 8; ++v) {
      const int row = v + mrow;
      sV1[row * LDPAD + t0 * 16 + ncol] = (__bf16)fast_tanhf(acc0[v]);
      sV1[row * LDPAD + t1 * 16 + ncol] = (__bf16)fast_tanhf(acc1[v]);
    }
    if (wave == 0 && ncol < EE) {
#pragma unroll
      for (int v = 0; v < 8; ++v) sLog[v + mrow][ncol] = accG[v];
    }
  }
  __syncthreads();

  // ---- Phase 2: softmax over experts (4) per row ---------------------------
  if (threadIdx.x < 16) {
    const int r = threadIdx.x;
    float l0 = sLog[r][0], l1 = sLog[r][1], l2 = sLog[r][2], l3 = sLog[r][3];
    float m = fmaxf(fmaxf(l0, l1), fmaxf(l2, l3));
    float e0 = __expf(l0 - m), e1 = __expf(l1 - m);
    float e2 = __expf(l2 - m), e3 = __expf(l3 - m);
    float inv = 1.0f / (e0 + e1 + e2 + e3);
    sG[r][0] = e0 * inv; sG[r][1] = e1 * inv;
    sG[r][2] = e2 * inv; sG[r][3] = e3 * inv;
  }
  __syncthreads();

  // ---- Phase 3: v2 = tanh(C_e @ v1), scale by gate -> w --------------------
#pragma unroll
  for (int tt = 0; tt < 2; ++tt) {
    const int t = wave * 2 + tt;  // 0..15 : e = t>>2, r-tile nt = t&3
    const int e = t >> 2;
    const int nt = t & 3;
    // issue all fragment loads, then both wmmas
    v16bf a0 = load_a_lds(sV1, LDPAD, e * RR, lane);
    v16bf a1 = load_a_lds(sV1, LDPAD, e * RR + 32, lane);
    v16bf c0 = load_b_tr(Ct + e * RR * RR, RR, nt * 16, 0, lane);
    v16bf c1 = load_b_tr(Ct + e * RR * RR, RR, nt * 16, 32, lane);
    v8f acc{};
    acc = wmma_bf16(a0, c0, acc);
    acc = wmma_bf16(a1, c1, acc);
#pragma unroll
    for (int v = 0; v < 8; ++v) {
      const int row = v + mrow;
      const float g = sG[row][e];
      sW[row * LDPAD + e * RR + nt * 16 + ncol] = (__bf16)(g * fast_tanhf(acc[v]));
    }
  }
  __syncthreads();

  // ---- Phase 4: GEMM2 (w @ U^T) + fused epilogue ---------------------------
  {
    v8f acc[8] = {};
    v16bf a = load_a_lds(sW, LDPAD, 0, lane);
#pragma unroll
    for (int kk = 0; kk < 8; ++kk) {
      // clause all 8 B loads so waits stagger across the 8 wmmas
      v16bf b[8];
#pragma unroll
      for (int j = 0; j < 8; ++j)
        b[j] = load_b_tr(Ut, EE * RR, (wave + 8 * j) * 16, kk * 32, lane);
      v16bf an = a;
      if (kk < 7) an = load_a_lds(sW, LDPAD, (kk + 1) * 32, lane);
#pragma unroll
      for (int j = 0; j < 8; ++j) acc[j] = wmma_bf16(a, b[j], acc[j]);
      a = an;
    }
#pragma unroll
    for (int j = 0; j < 8; ++j) {
      const int col = (wave + 8 * j) * 16 + ncol;
      const float bcol = bias[col];
#pragma unroll
      for (int v = 0; v < 8; ++v) {
        const int row = rowbase + v + mrow;
        const float xl = xin[row * DD + col];
        const float xz = x0[row * DD + col];
        xout[row * DD + col] = fmaf(xz, acc[j][v] + bcol, xl);
      }
    }
  }
}

// ---------------------------------------------------------------------------
// Host launch: prep weights into ws, then 3 fused layer launches.
// Layer 0: d_in -> d_out; layers 1,2: in-place on d_out (block-private rows,
// all x_l reads precede epilogue writes -> race-free).
// ---------------------------------------------------------------------------
extern "C" void kernel_launch(void* const* d_in, const int* in_sizes, int n_in,
                              void* d_out, int out_size, void* d_ws, size_t ws_size,
                              hipStream_t stream) {
  (void)in_sizes; (void)n_in; (void)out_size; (void)ws_size;
  const float* x0   = (const float*)d_in[0];
  const float* U    = (const float*)d_in[1];
  const float* V    = (const float*)d_in[2];
  const float* C    = (const float*)d_in[3];
  const float* gw   = (const float*)d_in[4];
  const float* bias = (const float*)d_in[5];
  float* out = (float*)d_out;

  const size_t nUV = (size_t)LL * EE * DD * RR;   // 786432
  __bf16* Vt = (__bf16*)d_ws;                     // [L][256][1024]
  __bf16* Ut = Vt + nUV;                          // [L][1024][256]
  __bf16* Ct = Ut + nUV;                          // [L][4][64][64]
  __bf16* Gt = Ct + (size_t)LL * EE * RR * RR;    // [16][1024]

  crossnet_prep_kernel<<<(int)((nUV + 255) / 256), 256, 0, stream>>>(
      U, V, C, gw, Vt, Ut, Ct, Gt);

  const int nblocks = BB / 16;  // 1024
  for (int l = 0; l < LL; ++l) {
    const float* xin = (l == 0) ? x0 : out;
    crossnet_layer_kernel<<<nblocks, 256, 0, stream>>>(
        xin, x0,
        Vt + (size_t)l * EE * RR * DD,
        Gt,
        Ct + (size_t)l * EE * RR * RR,
        Ut + (size_t)l * DD * EE * RR,
        bias + (size_t)l * DD,
        out);
  }
}
